// CustomSelfAttention_23063974379712
// MI455X (gfx1250) — matrix operand order repaired and checked by reading, not compiled, not run on hardware
//
#include <hip/hip_runtime.h>

// ---------------------------------------------------------------------------
// CDNA5 (gfx1250) self-attention: bf16 WMMA (f32 accumulate) throughout.
// Projection GEMMs use 64x64 register tiles per wave (16 WMMA per 16 b128
// loads) with a relaxed VGPR budget (waves_per_eu) so accumulators stay
// in registers; all WMMA fragments are fed by 128-bit loads.
// ---------------------------------------------------------------------------

typedef __attribute__((ext_vector_type(16))) __bf16 v16bf;
typedef __attribute__((ext_vector_type(8)))  __bf16 v8bf;
typedef __attribute__((ext_vector_type(4)))  __bf16 v4bf;
typedef __attribute__((ext_vector_type(4)))  float  v4f;
typedef __attribute__((ext_vector_type(8)))  float  v8f;

namespace {
constexpr int BB = 8;            // batch
constexpr int SS = 1024;         // sequence
constexpr int DD = 1024;         // model dim
constexpr int HH = 16;           // heads
constexpr int KD = 64;           // key dim
constexpr int HK = HH * KD;      // 1024
constexpr int BS = BB * SS;      // 8192
}

#define WMMA_BF16(A, B, C) \
  __builtin_amdgcn_wmma_f32_16x16x32_bf16(false, (A), false, (B), (short)0, (C), false, false)

__device__ __forceinline__ __bf16 f32_to_bf16(float f) {
  unsigned u = __builtin_bit_cast(unsigned, f);
  unsigned r = u + 0x7FFFu + ((u >> 16) & 1u);   // round-to-nearest-even
  unsigned short h = (unsigned short)(r >> 16);
  return __builtin_bit_cast(__bf16, h);
}

__device__ __forceinline__ v16bf combine16(v8bf lo, v8bf hi) {
  return __builtin_shufflevector(lo, hi, 0, 1, 2, 3, 4, 5, 6, 7,
                                 8, 9, 10, 11, 12, 13, 14, 15);
}

// ---- WMMA fragment loaders (layouts per CDNA5 ISA, wave32) -----
// All produce two 128-bit loads; caller guarantees 16B alignment.

// A (16x32 bf16): lane grp 0: elems 0..7 -> k 0..7, 8..15 -> k 16..23
//                 lane grp 1: elems 0..7 -> k 8..15, 8..15 -> k 24..31
__device__ __forceinline__ v16bf load_a_frag(const __bf16* A, int lda, int lane) {
  int row = lane & 15;
  int grp = lane >> 4;
  const __bf16* p = A + (size_t)row * lda + 8 * grp;
  return combine16(*(const v8bf*)p, *(const v8bf*)(p + 16));
}

// B (32x16 bf16) from k-contiguous (transposed) memory: BT[col][k], leading
// dim ldt along k. elems 0..7 -> k 8*grp+e, 8..15 -> k 16+8*grp+(e-8).
__device__ __forceinline__ v16bf load_bT_frag(const __bf16* BT, int ldt, int lane) {
  int col = lane & 15;
  int grp = lane >> 4;
  const __bf16* p = BT + (size_t)col * ldt + 8 * grp;
  return combine16(*(const v8bf*)p, *(const v8bf*)(p + 16));
}

// Row reductions across the 16-lane group that shares one C-matrix row.
__device__ __forceinline__ float row_max16(float v) {
#pragma unroll
  for (int off = 1; off < 16; off <<= 1) v = fmaxf(v, __shfl_xor(v, off, 32));
  return v;
}
__device__ __forceinline__ float row_sum16(float v) {
#pragma unroll
  for (int off = 1; off < 16; off <<= 1) v += __shfl_xor(v, off, 32);
  return v;
}

// ---------------------------------------------------------------------------
// Phase 0a: f32 -> bf16 conversion (4 elements / thread, vector ld/st)
// ---------------------------------------------------------------------------
__global__ void cvt_f32_bf16(const float* __restrict__ src,
                             __bf16* __restrict__ dst, int n4) {
  int i = blockIdx.x * blockDim.x + threadIdx.x;
  if (i >= n4) return;
  v4f f = *(const v4f*)(src + 4 * (size_t)i);
  v4bf o;
#pragma unroll
  for (int e = 0; e < 4; ++e) o[e] = f32_to_bf16(f[e]);
  *(v4bf*)(dst + 4 * (size_t)i) = o;
}

// ---------------------------------------------------------------------------
// Phase 0b: f32 [rows][cols] -> bf16 transposed [cols][rows]
// (one-time 2MB/matrix pass; coalesced writes)
// ---------------------------------------------------------------------------
__global__ void cvt_transpose_f32_bf16(const float* __restrict__ src,
                                       __bf16* __restrict__ dst,
                                       int rows, int cols) {
  int i = blockIdx.x * blockDim.x + threadIdx.x;
  if (i >= rows * cols) return;
  int n = i / rows;   // output row  (source column)
  int d = i % rows;   // output col  (source row)
  dst[i] = f32_to_bf16(src[(size_t)d * cols + n]);
}

// ---------------------------------------------------------------------------
// Phase 1: fused QKV projection. 64x64 register tile per wave, 8 waves
// (2 in M x 4 in N) -> 128x256 per block. grid = (BS/128, HK/256, 3).
// B operand = transposed weights [HK][D].  Q scaled by 1/sqrt(K)=0.125.
// Q,K stored [B,H,S,64]; V stored transposed [B,H,64,S] for the P*V GEMM.
// ---------------------------------------------------------------------------
__global__ __launch_bounds__(256)
__attribute__((amdgpu_waves_per_eu(1, 4)))
void qkv_gemm_kernel(const __bf16* __restrict__ xb,
                     const __bf16* __restrict__ wqT,
                     const __bf16* __restrict__ wkT,
                     const __bf16* __restrict__ wvT,
                     const float* __restrict__ bq,
                     const float* __restrict__ bk,
                     const float* __restrict__ bv,
                     __bf16* __restrict__ qo,
                     __bf16* __restrict__ ko,
                     __bf16* __restrict__ vo) {
  int lane = threadIdx.x & 31;
  int wave = threadIdx.x >> 5;   // 0..7
  int wm = wave & 1;             // 2 waves along M
  int wn = wave >> 1;            // 4 waves along N
  int m0 = blockIdx.x * 128 + wm * 64;
  int n0 = blockIdx.y * 256 + wn * 64;
  int which = blockIdx.z;
  const __bf16* wT  = which == 0 ? wqT : (which == 1 ? wkT : wvT);
  const float* bias = which == 0 ? bq : (which == 1 ? bk : bv);
  float scale       = which == 0 ? 0.125f : 1.0f;

  const __bf16* aptr = xb + (size_t)m0 * DD;
  const __bf16* bptr = wT + (size_t)n0 * DD;   // row n is contiguous in k
  v8f acc[4][4] = {};
#pragma unroll 1
  for (int k0 = 0; k0 < DD; k0 += 32) {
    v16bf a[4], b[4];
#pragma unroll
    for (int i = 0; i < 4; ++i) {
      a[i] = load_a_frag(aptr + (size_t)(16 * i) * DD + k0, DD, lane);
      b[i] = load_bT_frag(bptr + (size_t)(16 * i) * DD + k0, DD, lane);
    }
#pragma unroll
    for (int i = 0; i < 4; ++i)
#pragma unroll
      for (int j = 0; j < 4; ++j)
        acc[i][j] = WMMA_BF16(a[i], b[j], acc[i][j]);
  }

  int col = lane & 15, grp = lane >> 4;
#pragma unroll
  for (int j = 0; j < 4; ++j) {
    int gn = n0 + j * 16 + col;
    int h = gn >> 6, kk = gn & 63;
    float bv2 = bias[gn];
#pragma unroll
    for (int i = 0; i < 4; ++i) {
#pragma unroll
      for (int r = 0; r < 8; ++r) {
        int gm = m0 + i * 16 + r + 8 * grp;
        int bb = gm >> 10, s = gm & 1023;
        size_t bh = (size_t)(bb * HH + h);
        __bf16 val = f32_to_bf16((acc[i][j][r] + bv2) * scale);
        if (which == 0)      qo[(bh * SS + s) * KD + kk] = val;   // [B,H,S,64]
        else if (which == 1) ko[(bh * SS + s) * KD + kk] = val;   // [B,H,S,64]
        else                 vo[(bh * KD + kk) * SS + s] = val;   // [B,H,64,S]
      }
    }
  }
}

// ---------------------------------------------------------------------------
// Phase 2: flash attention. grid = (SS/64, H, B), block = 128 (4 waves);
// each wave owns one 16-query tile, streams keys 32 at a time.
// ---------------------------------------------------------------------------
__global__ void attn_kernel(const __bf16* __restrict__ qb,
                            const __bf16* __restrict__ kbuf,
                            const __bf16* __restrict__ vbufT,
                            const unsigned char* __restrict__ mask,
                            __bf16* __restrict__ ctxb) {
  int lane = threadIdx.x & 31;
  int wave = threadIdx.x >> 5;
  int b = blockIdx.z, h = blockIdx.y;
  int q0 = (blockIdx.x * 4 + wave) * 16;
  size_t bh = (size_t)(b * HH + h);
  const __bf16* Q  = qb    + (bh * SS + q0) * KD;
  const __bf16* Kt = kbuf  + bh * SS * KD;   // [S][64]
  const __bf16* Vt = vbufT + bh * KD * SS;   // [64][S]

  __shared__ __bf16 Plds[4][16][32];
  __bf16* P = &Plds[wave][0][0];

  int col = lane & 15, grp = lane >> 4;
  v16bf aq0 = load_a_frag(Q, KD, lane);       // kk 0..31
  v16bf aq1 = load_a_frag(Q + 32, KD, lane);  // kk 32..63

  float mrow[8], lrow[8];
  bool mq[8];
  v8f o[4] = {};
#pragma unroll
  for (int r = 0; r < 8; ++r) {
    mrow[r] = -3.0e38f;
    lrow[r] = 0.f;
    mq[r] = mask[b * SS + q0 + r + 8 * grp] != 0;
  }

  for (int kb0 = 0; kb0 < SS; kb0 += 32) {
    // scores: two 16x16 tiles (keys kb0..+15 and kb0+16..+31).
    // B fragment = K^T: for key column `col`, contraction dim kk is the
    // contiguous head-dim axis of [S][64] -> load_bT_frag with ldt=64.
    const __bf16* K0 = Kt + (size_t)kb0 * KD;
    const __bf16* K1 = Kt + (size_t)(kb0 + 16) * KD;
    v8f s0 = {}, s1 = {};
    s0 = WMMA_BF16(aq0, load_bT_frag(K0, KD, lane), s0);
    s0 = WMMA_BF16(aq1, load_bT_frag(K0 + 32, KD, lane), s0);
    s1 = WMMA_BF16(aq0, load_bT_frag(K1, KD, lane), s1);
    s1 = WMMA_BF16(aq1, load_bT_frag(K1 + 32, KD, lane), s1);

    bool km0 = mask[b * SS + kb0 + col] != 0;
    bool km1 = mask[b * SS + kb0 + 16 + col] != 0;
#pragma unroll
    for (int r = 0; r < 8; ++r) {
      float v0 = s0[r] + ((mq[r] && km0) ? 0.f : -1e9f);
      float v1 = s1[r] + ((mq[r] && km1) ? 0.f : -1e9f);
      float rmax = row_max16(fmaxf(v0, v1));
      float nm = fmaxf(mrow[r], rmax);
      float alpha = __expf(mrow[r] - nm);
      float p0 = __expf(v0 - nm);
      float p1 = __expf(v1 - nm);
      lrow[r] = lrow[r] * alpha + row_sum16(p0 + p1);
      mrow[r] = nm;
#pragma unroll
      for (int t = 0; t < 4; ++t) o[t][r] *= alpha;
      // stage P (C layout -> LDS row-major); wave-private, LDS is in-order
      P[(r + 8 * grp) * 32 + col]      = f32_to_bf16(p0);
      P[(r + 8 * grp) * 32 + col + 16] = f32_to_bf16(p1);
    }
    // P fragment: row-major 16x32 in LDS -> plain A-fragment loads (b128)
    v16bf ap = load_a_frag(P, 32, lane);
#pragma unroll
    for (int t = 0; t < 4; ++t) {
      // V^T [64][S]: for output column t*16+col, contraction (keys) is the
      // contiguous S axis -> load_bT_frag with ldt=SS.
      const __bf16* vtile = Vt + (size_t)(t * 16) * SS + kb0;
      o[t] = WMMA_BF16(ap, load_bT_frag(vtile, SS, lane), o[t]);
    }
  }

  // normalize, write ctx bf16 in [B*S, H*K] row-major for the output GEMM
#pragma unroll
  for (int r = 0; r < 8; ++r) {
    float inv = 1.0f / lrow[r];
    int gq = q0 + r + 8 * grp;
    size_t base = ((size_t)(b * SS + gq)) * HK + h * KD;
#pragma unroll
    for (int t = 0; t < 4; ++t)
      ctxb[base + t * 16 + col] = f32_to_bf16(o[t][r] * inv);
  }
}

// ---------------------------------------------------------------------------
// Phase 3: output projection. 64x64 register tile per wave, 8 waves
// (2 in M x 4 in N) -> 128x256 per block. grid = (BS/128, D/256).
// B operand = transposed Wo [D][HK].
// ---------------------------------------------------------------------------
__global__ __launch_bounds__(256)
__attribute__((amdgpu_waves_per_eu(1, 4)))
void out_gemm_kernel(const __bf16* __restrict__ ctxb,
                     const __bf16* __restrict__ woT,
                     const float* __restrict__ bo,
                     float* __restrict__ out) {
  int lane = threadIdx.x & 31;
  int wave = threadIdx.x >> 5;
  int wm = wave & 1;
  int wn = wave >> 1;
  int m0 = blockIdx.x * 128 + wm * 64;
  int n0 = blockIdx.y * 256 + wn * 64;

  const __bf16* aptr = ctxb + (size_t)m0 * HK;
  const __bf16* bptr = woT + (size_t)n0 * HK;
  v8f acc[4][4] = {};
#pragma unroll 1
  for (int k0 = 0; k0 < HK; k0 += 32) {
    v16bf a[4], b[4];
#pragma unroll
    for (int i = 0; i < 4; ++i) {
      a[i] = load_a_frag(aptr + (size_t)(16 * i) * HK + k0, HK, lane);
      b[i] = load_bT_frag(bptr + (size_t)(16 * i) * HK + k0, HK, lane);
    }
#pragma unroll
    for (int i = 0; i < 4; ++i)
#pragma unroll
      for (int j = 0; j < 4; ++j)
        acc[i][j] = WMMA_BF16(a[i], b[j], acc[i][j]);
  }

  int col = lane & 15, grp = lane >> 4;
#pragma unroll
  for (int j = 0; j < 4; ++j) {
    int gn = n0 + j * 16 + col;
    float bias = bo[gn];
#pragma unroll
    for (int i = 0; i < 4; ++i)
#pragma unroll
      for (int r = 0; r < 8; ++r)
        out[(size_t)(m0 + i * 16 + r + 8 * grp) * DD + gn] = acc[i][j][r] + bias;
  }
}

// ---------------------------------------------------------------------------
extern "C" void kernel_launch(void* const* d_in, const int* in_sizes, int n_in,
                              void* d_out, int out_size, void* d_ws, size_t ws_size,
                              hipStream_t stream) {
  const float* x            = (const float*)d_in[0];
  const unsigned char* mask = (const unsigned char*)d_in[1];
  const float* Wq = (const float*)d_in[2];
  const float* bq = (const float*)d_in[3];
  const float* Wk = (const float*)d_in[4];
  const float* bk = (const float*)d_in[5];
  const float* Wv = (const float*)d_in[6];
  const float* bv = (const float*)d_in[7];
  const float* Wo = (const float*)d_in[8];
  const float* bo = (const float*)d_in[9];
  float* out = (float*)d_out;

  // workspace carve-up (all bf16)
  char* ws = (char*)d_ws;
  size_t off = 0;
  auto carve = [&](size_t elems) {
    __bf16* p = (__bf16*)(ws + off);
    off += elems * sizeof(__bf16);
    return p;
  };
  __bf16* xb   = carve((size_t)BS * DD);            // 16 MB  [BS][D]
  __bf16* wqT  = carve((size_t)HK * DD);            //  2 MB  [HK][D]
  __bf16* wkT  = carve((size_t)HK * DD);
  __bf16* wvT  = carve((size_t)HK * DD);
  __bf16* woT  = carve((size_t)DD * HK);            //  2 MB  [D][HK]
  __bf16* qb   = carve((size_t)BB * HH * SS * KD);  // 16 MB  [B,H,S,64]
  __bf16* kb   = carve((size_t)BB * HH * SS * KD);  // 16 MB  [B,H,S,64]
  __bf16* vbT  = carve((size_t)BB * HH * KD * SS);  // 16 MB  [B,H,64,S]
  __bf16* ctxb = carve((size_t)BS * HK);            // 16 MB  [BS][HK]

  // Phase 0: convert x to bf16; convert+transpose weights
  {
    int n4 = (BS * DD) / 4;
    cvt_f32_bf16<<<(n4 + 255) / 256, 256, 0, stream>>>(x, xb, n4);
    int nt = DD * HK;
    cvt_transpose_f32_bf16<<<(nt + 255) / 256, 256, 0, stream>>>(Wq, wqT, DD, HK);
    cvt_transpose_f32_bf16<<<(nt + 255) / 256, 256, 0, stream>>>(Wk, wkT, DD, HK);
    cvt_transpose_f32_bf16<<<(nt + 255) / 256, 256, 0, stream>>>(Wv, wvT, DD, HK);
    // Wo is [HK][D]; we want [D][HK]: rows=HK, cols=D transposed
    cvt_transpose_f32_bf16<<<(nt + 255) / 256, 256, 0, stream>>>(Wo, woT, HK, DD);
  }
  // Phase 1: QKV projections (Q pre-scaled by 1/8)
  qkv_gemm_kernel<<<dim3(BS / 128, HK / 256, 3), 256, 0, stream>>>(
      xb, wqT, wkT, wvT, bq, bk, bv, qb, kb, vbT);
  // Phase 2: masked flash attention
  attn_kernel<<<dim3(SS / 64, HH, BB), 128, 0, stream>>>(qb, kb, vbT, mask, ctxb);
  // Phase 3: output projection
  out_gemm_kernel<<<dim3(BS / 128, DD / 256), 256, 0, stream>>>(ctxb, woT, bo, out);
}
